// EEG_GNN_36369783063171
// MI455X (gfx1250) — matrix operand order repaired and checked by reading, not compiled, not run on hardware
//
#include <hip/hip_runtime.h>
#include <hip/hip_bf16.h>
#include <math.h>

// ---------------- problem constants ----------------
#define NN      32768
#define L_IN    3000
#define PER_G   64
#define NUM_G   (NN / PER_G)        // 512
#define DEG     16
#define EDGES   (NN * DEG)          // 524288
#define E_PER_G (PER_G * DEG)       // 1024
#define HID     128
#define OUT3    64
#define NCLS    2
#define RED     146
#define F_IN    (32 * RED)          // 4672
#define KCONV   128                 // 100 padded to 128
#define BN_EPS  1e-5f

typedef __attribute__((ext_vector_type(16))) _Float16 v16h;
typedef __attribute__((ext_vector_type(8)))  _Float16 v8h;
typedef __attribute__((ext_vector_type(8)))  float    v8f;

union AB { v16h v; v8h h[2]; unsigned u[8]; };
union H2 { _Float16 h[2]; unsigned u; };

// ---------------- weight prep kernels ----------------
// W [K x C] f32 row-major  ->  Wt [C x K] f16
__global__ void w_transpose_f16(const float* __restrict__ W, _Float16* __restrict__ Wt,
                                int K, int C) {
  int idx = blockIdx.x * 256 + threadIdx.x;
  if (idx < K * C) {
    int k = idx / C, c = idx % C;
    Wt[(size_t)c * K + k] = (_Float16)W[idx];
  }
}

// conv_w (32,1,100) -> Wct [32][128] f16, K zero-padded 100->128
__global__ void conv_w_pack(const float* __restrict__ cw, _Float16* __restrict__ Wct) {
  int idx = blockIdx.x * 256 + threadIdx.x;
  if (idx < 32 * KCONV) {
    int c = idx >> 7, k = idx & 127;
    Wct[idx] = (k < 100) ? (_Float16)cw[c * 100 + k] : (_Float16)0.f;
  }
}

// ---------------- degree / norm (shared across all 3 GCN layers) ----------------
__global__ void deg_norm_kernel(const int* __restrict__ srcI, const int* __restrict__ dstI,
                                const float* __restrict__ w,
                                float* __restrict__ selfw, float* __restrict__ norm) {
  __shared__ float degs[PER_G];
  __shared__ float sdinv[PER_G];
  int g = blockIdx.x, tid = threadIdx.x;
  if (tid < PER_G) degs[tid] = 1.0f;            // self loop weight
  __syncthreads();
  int e0 = g * E_PER_G;
  for (int e = tid; e < E_PER_G; e += 256)
    atomicAdd(&degs[dstI[e0 + e] - g * PER_G], w[e0 + e]);
  __syncthreads();
  if (tid < PER_G) {
    float dv = rsqrtf(degs[tid]);
    sdinv[tid] = dv;
    selfw[g * PER_G + tid] = dv * dv;
  }
  __syncthreads();
  for (int e = tid; e < E_PER_G; e += 256) {
    int s = srcI[e0 + e] - g * PER_G;
    int d = dstI[e0 + e] - g * PER_G;
    norm[e0 + e] = sdinv[s] * w[e0 + e] * sdinv[d];
  }
}

// ---------------- Conv1d (k=100, stride=20) as WMMA GEMM per node ----------------
// output h0 f16 [N, 4672], feature index = c*146 + r   (ReLU(conv + b))
__global__ void conv_relu_wmma(const float* __restrict__ x,
                               const _Float16* __restrict__ Wct,   // [32][128]
                               const float* __restrict__ conv_b,
                               _Float16* __restrict__ h0) {
  __shared__ __align__(16) _Float16 xs[3040];          // zero-padded signal
  __shared__ __align__(16) _Float16 ws[32 * KCONV];    // weights
  int n = blockIdx.x;
  int tid = threadIdx.x;                               // 256 threads = 8 waves
  const float* xr = x + (size_t)n * L_IN;
  for (int i = tid; i < 3040; i += 256)
    xs[i] = (i < L_IN) ? (_Float16)xr[i] : (_Float16)0.f;
  for (int i = tid; i < 32 * KCONV; i += 256) ws[i] = Wct[i];
  __syncthreads();

  int wv = tid >> 5, lane = tid & 31;
  int l15 = lane & 15, hi = lane >> 4;
  const unsigned* xu = (const unsigned*)xs;

  // 10 m-tiles (146 rows padded to 160) x 2 n-tiles (32 channels) = 20 tiles / 8 waves
  for (int t = wv; t < 20; t += 8) {
    int mt = t % 10, nt = t / 10;
    int rbase = mt * 16 + l15;          // reduced position (A row)
    int cidx  = nt * 16 + l15;          // channel (B column)
    v8f acc = {};
    #pragma unroll
    for (int k0 = 0; k0 < KCONV; k0 += 32) {
      AB a, b;
      int xbase = rbase * 20 + k0 + hi * 8;      // even -> u32 aligned
      #pragma unroll
      for (int i = 0; i < 4; i++) a.u[i]     = xu[(xbase >> 1) + i];
      #pragma unroll
      for (int i = 0; i < 4; i++) a.u[4 + i] = xu[((xbase + 16) >> 1) + i];
      const unsigned* wu = (const unsigned*)(ws + cidx * KCONV + k0 + hi * 8);
      #pragma unroll
      for (int i = 0; i < 8; i++) b.u[i] = wu[i + (i & 4)];
      acc = __builtin_amdgcn_wmma_f32_16x16x32_f16(false, a.v, false, b.v,
                                                   (short)0, acc, false, false);
    }
    float bias = conv_b[cidx];
    _Float16* dst = h0 + (size_t)n * F_IN + (size_t)cidx * RED;
    int rp0 = mt * 16 + hi * 8;                        // even -> b32-aligned halves
    if (mt < 9) {
      // full tile: 8 contiguous halves -> 4 packed b32 stores, no branches
      #pragma unroll
      for (int i = 0; i < 4; i++) {
        H2 pk;
        float v0 = acc[2 * i]     + bias;
        float v1 = acc[2 * i + 1] + bias;
        pk.h[0] = (_Float16)(v0 > 0.f ? v0 : 0.f);
        pk.h[1] = (_Float16)(v1 > 0.f ? v1 : 0.f);
        *(unsigned*)(dst + rp0 + 2 * i) = pk.u;
      }
    } else if (hi == 0) {
      // tail tile: only rows 144,145 valid
      H2 pk;
      float v0 = acc[0] + bias;
      float v1 = acc[1] + bias;
      pk.h[0] = (_Float16)(v0 > 0.f ? v0 : 0.f);
      pk.h[1] = (_Float16)(v1 > 0.f ? v1 : 0.f);
      *(unsigned*)(dst + 144) = pk.u;
    }
  }
}

// ---------------- f16 WMMA GEMM, 4 n-tiles per wave ----------------
// Co[M,Nc] = A[M,K] * Bt[Nc,K]^T ; grid = (Nc/64, M/16), 1 wave/block.
// A fragment loaded once per K-step and reused by 4 WMMAs.
__global__ void __launch_bounds__(32)
wmma_gemm_f16_4n(const _Float16* __restrict__ A, const _Float16* __restrict__ Bt,
                 float* __restrict__ Co, int Nc, int K) {
  int ntb = blockIdx.x * 4, mt = blockIdx.y;
  int lane = threadIdx.x & 31;
  int l15 = lane & 15, hi = lane >> 4;
  size_t arow = (size_t)(mt * 16 + l15) * K;
  const _Float16* brow[4];
  #pragma unroll
  for (int j = 0; j < 4; j++)
    brow[j] = Bt + (size_t)((ntb + j) * 16 + l15) * K;
  v8f acc[4] = {{}, {}, {}, {}};
  for (int k0 = 0; k0 < K; k0 += 32) {
    AB a;
    a.h[0] = *(const v8h*)(A + arow + k0 + hi * 8);        // A: K=0..7 / 8..15
    a.h[1] = *(const v8h*)(A + arow + k0 + hi * 8 + 16);   //    K=16..23 / 24..31
    int bo = k0 + hi * 8;                                   // B: K=0..7,16..23 / 8..15,24..31
    #pragma unroll
    for (int j = 0; j < 4; j++) {
      AB b;
      b.h[0] = *(const v8h*)(brow[j] + bo);
      b.h[1] = *(const v8h*)(brow[j] + bo + 16);
      acc[j] = __builtin_amdgcn_wmma_f32_16x16x32_f16(false, a.v, false, b.v,
                                                      (short)0, acc[j], false, false);
    }
  }
  size_t rbase = (size_t)(mt * 16 + hi * 8) * Nc;
  #pragma unroll
  for (int j = 0; j < 4; j++) {
    size_t ob = rbase + (ntb + j) * 16 + l15;
    #pragma unroll
    for (int e = 0; e < 8; e++) Co[ob + (size_t)e * Nc] = acc[j][e];
  }
}

// ---------------- GCN aggregation + self loop + bias + BN + ReLU (per graph) ----------------
// one block per (graph, 64-channel slice); z slice + acc + staged edges all in LDS
template <int C, bool W16, bool W32>
__global__ void gcn_post_kernel(const float* __restrict__ z, const float* __restrict__ norm,
                                const float* __restrict__ selfw,
                                const int* __restrict__ srcI, const int* __restrict__ dstI,
                                const float* __restrict__ bias,
                                const float* __restrict__ gam, const float* __restrict__ bet,
                                const float* __restrict__ mean, const float* __restrict__ var,
                                _Float16* __restrict__ out16, float* __restrict__ out32) {
  __shared__ float zl[PER_G * 64];
  __shared__ float acc[PER_G * 64];
  __shared__ int   es[E_PER_G];
  __shared__ int   ed[E_PER_G];
  __shared__ float en[E_PER_G];
  int g = blockIdx.x, tid = threadIdx.x;          // 256 threads
  int cbase = blockIdx.y * 64;
  for (int i = tid; i < PER_G * 64; i += 256) {
    int nloc = i >> 6, cc = i & 63;
    zl[i] = z[((size_t)g * PER_G + nloc) * C + cbase + cc];
    acc[i] = 0.f;
  }
  int e0 = g * E_PER_G;
  for (int e = tid; e < E_PER_G; e += 256) {
    es[e] = srcI[e0 + e] - g * PER_G;
    ed[e] = dstI[e0 + e] - g * PER_G;
    en[e] = norm[e0 + e];
  }
  __syncthreads();
  int cc = tid & 63, el = tid >> 6;               // 4 edge lanes x 64 channels
  for (int e = el; e < E_PER_G; e += 4)
    atomicAdd(&acc[ed[e] * 64 + cc], zl[es[e] * 64 + cc] * en[e]);
  __syncthreads();
  for (int i = tid; i < PER_G * 64; i += 256) {
    int nloc = i >> 6, cc2 = i & 63;
    int c = cbase + cc2;
    float v = acc[i] + zl[i] * selfw[g * PER_G + nloc] + bias[c];
    v = (v - mean[c]) * rsqrtf(var[c] + BN_EPS) * gam[c] + bet[c];
    v = fmaxf(v, 0.f);
    size_t idx = ((size_t)g * PER_G + nloc) * C + c;
    if (W16) out16[idx] = (_Float16)v;
    if (W32) out32[idx] = v;
  }
}

// ---------------- global mean pool + FC + log_softmax ----------------
__global__ void pool_fc_kernel(const float* __restrict__ h3, const float* __restrict__ fc_w,
                               const float* __restrict__ fc_b, float* __restrict__ out) {
  __shared__ float pooled[OUT3];
  __shared__ float lg[NCLS];
  int g = blockIdx.x, tid = threadIdx.x;          // 64 threads
  float s = 0.f;
  for (int n = 0; n < PER_G; n++)
    s += h3[((size_t)g * PER_G + n) * OUT3 + tid];
  pooled[tid] = s * (1.f / (float)PER_G);
  __syncthreads();
  if (tid < NCLS) {
    float l = fc_b[tid];
    for (int c = 0; c < OUT3; c++) l += pooled[c] * fc_w[c * NCLS + tid];
    lg[tid] = l;
  }
  __syncthreads();
  if (tid < NCLS) {
    float m = fmaxf(lg[0], lg[1]);
    float lse = m + logf(expf(lg[0] - m) + expf(lg[1] - m));
    out[g * NCLS + tid] = lg[tid] - lse;
  }
}

// ---------------- launcher ----------------
static inline size_t align256(size_t x) { return (x + 255) & ~(size_t)255; }

extern "C" void kernel_launch(void* const* d_in, const int* in_sizes, int n_in,
                              void* d_out, int out_size, void* d_ws, size_t ws_size,
                              hipStream_t stream) {
  const float* x        = (const float*)d_in[0];
  const int*   eidx     = (const int*)d_in[1];
  const int*   srcI     = eidx;
  const int*   dstI     = eidx + EDGES;
  const float* eattr    = (const float*)d_in[2];
  // d_in[3] = batch (layout is known: g = n/64, unused)
  const float* conv_w   = (const float*)d_in[4];
  const float* conv_b   = (const float*)d_in[5];
  const float* W1       = (const float*)d_in[6];
  const float* b1       = (const float*)d_in[7];
  const float* W2       = (const float*)d_in[8];
  const float* b2       = (const float*)d_in[9];
  const float* W3       = (const float*)d_in[10];
  const float* b3       = (const float*)d_in[11];
  const float* g1  = (const float*)d_in[12]; const float* be1 = (const float*)d_in[13];
  const float* m1  = (const float*)d_in[14]; const float* v1  = (const float*)d_in[15];
  const float* g2  = (const float*)d_in[16]; const float* be2 = (const float*)d_in[17];
  const float* m2  = (const float*)d_in[18]; const float* v2  = (const float*)d_in[19];
  const float* g3  = (const float*)d_in[20]; const float* be3 = (const float*)d_in[21];
  const float* m3  = (const float*)d_in[22]; const float* v3  = (const float*)d_in[23];
  const float* fc_w = (const float*)d_in[24]; const float* fc_b = (const float*)d_in[25];
  float* out = (float*)d_out;

  // workspace carve-up
  char* p = (char*)d_ws;
  size_t off = 0;
  _Float16* W1t  = (_Float16*)(p + off); off = align256(off + (size_t)HID * F_IN * 2);
  _Float16* W2t  = (_Float16*)(p + off); off = align256(off + (size_t)HID * HID * 2);
  _Float16* W3t  = (_Float16*)(p + off); off = align256(off + (size_t)OUT3 * HID * 2);
  _Float16* Wct  = (_Float16*)(p + off); off = align256(off + (size_t)32 * KCONV * 2);
  float*    selfw= (float*)(p + off);    off = align256(off + (size_t)NN * 4);
  float*    nrm  = (float*)(p + off);    off = align256(off + (size_t)EDGES * 4);
  _Float16* h0   = (_Float16*)(p + off); off = align256(off + (size_t)NN * F_IN * 2);
  float*    z    = (float*)(p + off);    off = align256(off + (size_t)NN * HID * 4);
  _Float16* h16  = (_Float16*)(p + off); off = align256(off + (size_t)NN * HID * 2);
  float*    h3f  = (float*)(p + off);    off = align256(off + (size_t)NN * OUT3 * 4);
  (void)ws_size; (void)n_in; (void)in_sizes; (void)out_size;

  // 1. weight prep
  w_transpose_f16<<<(F_IN * HID + 255) / 256, 256, 0, stream>>>(W1, W1t, F_IN, HID);
  w_transpose_f16<<<(HID * HID + 255) / 256, 256, 0, stream>>>(W2, W2t, HID, HID);
  w_transpose_f16<<<(HID * OUT3 + 255) / 256, 256, 0, stream>>>(W3, W3t, HID, OUT3);
  conv_w_pack<<<(32 * KCONV + 255) / 256, 256, 0, stream>>>(conv_w, Wct);

  // 2. degree / norm (shared by all 3 GCN layers)
  deg_norm_kernel<<<NUM_G, 256, 0, stream>>>(srcI, dstI, eattr, selfw, nrm);

  // 3. Conv1d + ReLU -> h0 f16 [N, F_IN]
  conv_relu_wmma<<<NN, 256, 0, stream>>>(x, Wct, conv_b, h0);

  // 4. GCN1: z = h0 @ W1 ; post -> h16
  wmma_gemm_f16_4n<<<dim3(HID / 64, NN / 16), 32, 0, stream>>>(h0, W1t, z, HID, F_IN);
  gcn_post_kernel<HID, true, false><<<dim3(NUM_G, HID / 64), 256, 0, stream>>>(
      z, nrm, selfw, srcI, dstI, b1, g1, be1, m1, v1, h16, nullptr);

  // 5. GCN2
  wmma_gemm_f16_4n<<<dim3(HID / 64, NN / 16), 32, 0, stream>>>(h16, W2t, z, HID, HID);
  gcn_post_kernel<HID, true, false><<<dim3(NUM_G, HID / 64), 256, 0, stream>>>(
      z, nrm, selfw, srcI, dstI, b2, g2, be2, m2, v2, h16, nullptr);

  // 6. GCN3 (C=64)
  wmma_gemm_f16_4n<<<dim3(OUT3 / 64, NN / 16), 32, 0, stream>>>(h16, W3t, z, OUT3, HID);
  gcn_post_kernel<OUT3, false, true><<<dim3(NUM_G, 1), 256, 0, stream>>>(
      z, nrm, selfw, srcI, dstI, b3, g3, be3, m3, v3, nullptr, h3f);

  // 7. pool + FC + log_softmax -> out [512, 2]
  pool_fc_kernel<<<NUM_G, OUT3, 0, stream>>>(h3f, fc_w, fc_b, out);
}
